// LSTM_19645180412468
// MI455X (gfx1250) — compile-verified, matrix operand order checked by reading, not run on hardware
//
#include <hip/hip_runtime.h>
#include <hip/hip_bf16.h>

// ---------------------------------------------------------------------------
// 2-layer LSTM for MI455X (gfx1250): bf16 WMMA GEMM + fused cell update.
// B=64, T=512, D=512, H=1024.  Per step: z = [x_t, h] @ W^T + b (M=64,
// N=4096, K=1536/2048), then i,f,o,g nonlinearities and c/h update.
// Weights converted to bf16 and pre-packed into WMMA B-fragment tile order;
// both weight matrices (~29MB bf16) live in the 192MB L2 across all 512
// steps, so the recurrence streams weights from L2, not HBM.
// ---------------------------------------------------------------------------

typedef __bf16 bf16_t;
typedef __attribute__((ext_vector_type(16))) __bf16 v16bf;
typedef __attribute__((ext_vector_type(8)))  float  v8f;

#define LSTM_BATCH 64
#define LSTM_TLEN  512
#define LSTM_DIN0  512
#define LSTM_HDIM  1024

// ---- fp32 -> bf16 bulk convert -------------------------------------------
__global__ __launch_bounds__(256) void f32_to_bf16_kernel(
    const float* __restrict__ src, bf16_t* __restrict__ dst, long n) {
  long i = (long)blockIdx.x * 256 + threadIdx.x;
  if (i < n) dst[i] = (bf16_t)src[i];
}

// ---- zero-fill helpers ----------------------------------------------------
__global__ __launch_bounds__(256) void zero_f32_kernel(float* __restrict__ p, long n) {
  long i = (long)blockIdx.x * 256 + threadIdx.x;
  if (i < n) p[i] = 0.0f;
}
__global__ __launch_bounds__(256) void zero_bf16_kernel(bf16_t* __restrict__ p, long n) {
  long i = (long)blockIdx.x * 256 + threadIdx.x;
  if (i < n) p[i] = (bf16_t)0.0f;
}

// ---- pack W (4H x K, row-major fp32) into bf16 WMMA B-fragment tiles ------
// Tile (nt, kt) covers W rows nt*16..+15 (the output columns of the GEMM)
// and k columns kt*32..+31.  Within a tile, element index = lane*16 + j with
//   lane 0..15 : N = lane,      K = kt*32 + j        (j = 0..15)
//   lane 16..31: N = lane - 16, K = kt*32 + 16 + j
// so the step kernel's per-lane contiguous 32B load yields the B fragment.
__global__ __launch_bounds__(256) void pack_weights_kernel(
    const float* __restrict__ W, bf16_t* __restrict__ Wp, int K, long total) {
  long idx = (long)blockIdx.x * 256 + threadIdx.x;
  if (idx >= total) return;
  int  Kt   = K >> 5;
  int  j    = (int)(idx & 15);
  int  lane = (int)((idx >> 4) & 31);
  long tile = idx >> 9;
  long kt   = tile % Kt;
  long nt   = tile / Kt;
  long n    = nt * 16 + (lane & 15);
  long k    = kt * 32 + (long)(lane >> 4) * 16 + j;
  Wp[idx] = (bf16_t)W[n * (long)K + k];
}

__device__ __forceinline__ float sigmoidf_dev(float x) {
  return 1.0f / (1.0f + __expf(-x));
}

// ---- fused LSTM step ------------------------------------------------------
// Grid: H/16 = 64 blocks of 256 threads (8 waves, wave32).
// Block bn owns output columns [bn*16, bn*16+16) of h.
// Wave w: gate g = w>>1, m-tiles mt0 = (w&1)*2 and mt0+1  (M = 64 = 4 tiles).
// Each wave runs two v_wmma_f32_16x16x32_bf16 accumulators over K, dumps the
// fp32 D-tiles to LDS, then all threads do the fused cell update.
template <int DIN>
__global__ __launch_bounds__(256) void lstm_step_kernel(
    const bf16_t* __restrict__ xa, long sx,   // x-part rows: xa + m*sx, k in [0,DIN)
    const bf16_t* __restrict__ ha, long sh,   // h-part rows: ha + m*sh, k in [0,H)
    const bf16_t* __restrict__ Wp,            // packed bf16 weights
    const float*  __restrict__ bias,          // 4H fp32
    float*        __restrict__ cst,           // 64 x H fp32 cell state (in/out)
    bf16_t*       __restrict__ hout, long soh,// bf16 h_next rows
    float*        __restrict__ fout, long sof)// optional fp32 h_next rows
{
  constexpr int H  = LSTM_HDIM;
  constexpr int K  = DIN + H;
  constexpr int Kt = K / 32;

  __shared__ float zls[4][LSTM_BATCH][16];   // 16 KB: z tiles, fp32

  const int bn   = blockIdx.x;
  const int tid  = threadIdx.x;
  const int wave = tid >> 5;
  const int lane = tid & 31;
  const int g    = wave >> 1;           // gate 0..3 (i,f,o,g)
  const int mt0  = (wave & 1) * 2;      // first of two m-tiles
  const int lrow = lane & 15;           // A-fragment row within tile
  const int lko  = (lane >> 4) * 16;    // A-fragment k sub-offset

  v8f acc0 = {};
  v8f acc1 = {};

  const long nt = (long)g * (H / 16) + bn;                  // W row-tile index
  const bf16_t* wt = Wp + nt * (long)Kt * 512 + (long)lane * 16;

  // ---- K sweep, x segment [0, DIN) ----
  {
    const bf16_t* ap = xa + (long)(mt0 * 16 + lrow) * sx + lko;
    const bf16_t* aq = ap + 16 * sx;
#pragma unroll 4
    for (int kt = 0; kt < DIN / 32; ++kt) {
      v16bf a0 = *(const v16bf*)ap;
      v16bf a1 = *(const v16bf*)aq;
      v16bf bb = *(const v16bf*)wt;
      __builtin_prefetch((const void*)(wt + 512), 0, 1);    // global_prefetch_b8
      ap += 32; aq += 32; wt += 512;
      acc0 = __builtin_amdgcn_wmma_f32_16x16x32_bf16(false, a0, false, bb,
                                                     (short)0, acc0, false, false);
      acc1 = __builtin_amdgcn_wmma_f32_16x16x32_bf16(false, a1, false, bb,
                                                     (short)0, acc1, false, false);
    }
  }
  // ---- K sweep, h segment [DIN, DIN+H) ----
  {
    const bf16_t* ap = ha + (long)(mt0 * 16 + lrow) * sh + lko;
    const bf16_t* aq = ap + 16 * sh;
#pragma unroll 4
    for (int kt = 0; kt < H / 32; ++kt) {
      v16bf a0 = *(const v16bf*)ap;
      v16bf a1 = *(const v16bf*)aq;
      v16bf bb = *(const v16bf*)wt;
      __builtin_prefetch((const void*)(wt + 512), 0, 1);
      ap += 32; aq += 32; wt += 512;
      acc0 = __builtin_amdgcn_wmma_f32_16x16x32_bf16(false, a0, false, bb,
                                                     (short)0, acc0, false, false);
      acc1 = __builtin_amdgcn_wmma_f32_16x16x32_bf16(false, a1, false, bb,
                                                     (short)0, acc1, false, false);
    }
  }

  // ---- D-tile -> LDS. 16x16 f32 layout: VGPR r, lanes 0-15 -> M=r, N=lane;
  //      lanes 16-31 -> M=8+r, N=lane-16.
  const int mlh = (lane >> 4) * 8;
  const int nn  = lane & 15;
#pragma unroll
  for (int r = 0; r < 8; ++r) {
    zls[g][mt0 * 16 +      mlh + r][nn] = acc0[r];
    zls[g][mt0 * 16 + 16 + mlh + r][nn] = acc1[r];
  }
  __syncthreads();

  // ---- fused cell update: 64 rows x 16 cols = 1024 elems, 4 per thread ----
  for (int e = tid; e < LSTM_BATCH * 16; e += 256) {
    int m  = e >> 4;
    int n  = e & 15;
    int ng = bn * 16 + n;
    float zi = zls[0][m][n] + bias[0 * H + ng];
    float zf = zls[1][m][n] + bias[1 * H + ng];
    float zo = zls[2][m][n] + bias[2 * H + ng];
    float zg = zls[3][m][n] + bias[3 * H + ng];
    float ig = sigmoidf_dev(zi);
    float fg = sigmoidf_dev(zf);
    float og = sigmoidf_dev(zo);
    float gg = tanhf(zg);
    long  ci = (long)m * H + ng;
    float c  = fg * cst[ci] + ig * gg;
    cst[ci]  = c;
    float h  = og * tanhf(c);
    hout[(long)m * soh + ng] = (bf16_t)h;
    if (fout) fout[(long)m * sof + ng] = h;
  }
}

// ---------------------------------------------------------------------------
extern "C" void kernel_launch(void* const* d_in, const int* in_sizes, int n_in,
                              void* d_out, int out_size, void* d_ws, size_t ws_size,
                              hipStream_t stream) {
  (void)in_sizes; (void)n_in; (void)out_size; (void)ws_size;

  const float* x  = (const float*)d_in[0];   // (B,T,D)
  const float* W0 = (const float*)d_in[1];   // (4H, D+H)
  const float* b0 = (const float*)d_in[2];   // (4H,)
  const float* W1 = (const float*)d_in[3];   // (4H, 2H)
  const float* b1 = (const float*)d_in[4];   // (4H,)
  float* out = (float*)d_out;                // (B,T,H) fp32

  const long Bz = LSTM_BATCH, Tz = LSTM_TLEN, Dz = LSTM_DIN0, Hz = LSTM_HDIM;
  const long K0 = Dz + Hz;      // 1536
  const long K1 = Hz + Hz;      // 2048

  // ---- carve workspace (256B aligned) ----
  char* p = (char*)d_ws;
  auto carve = [&](size_t bytes) -> char* {
    char* r = p;
    p += (bytes + 255) & ~(size_t)255;
    return r;
  };
  bf16_t* xbf = (bf16_t*)carve((size_t)(Bz * Tz * Dz) * sizeof(bf16_t)); // 33.6 MB
  bf16_t* W0p = (bf16_t*)carve((size_t)(4 * Hz * K0) * sizeof(bf16_t));  // 12.6 MB
  bf16_t* W1p = (bf16_t*)carve((size_t)(4 * Hz * K1) * sizeof(bf16_t));  // 16.8 MB
  bf16_t* y0  = (bf16_t*)carve((size_t)(Bz * Tz * Hz) * sizeof(bf16_t)); // 67.1 MB
  bf16_t* h1a = (bf16_t*)carve((size_t)(Bz * Hz) * sizeof(bf16_t));
  bf16_t* h1b = (bf16_t*)carve((size_t)(Bz * Hz) * sizeof(bf16_t));
  bf16_t* zb  = (bf16_t*)carve((size_t)2048 * sizeof(bf16_t));           // zero h0
  float*  c0  = (float*)carve((size_t)(Bz * Hz) * sizeof(float));
  float*  c1  = (float*)carve((size_t)(Bz * Hz) * sizeof(float));

  // ---- one-time (per launch) conversion / packing / state init ----
  {
    long nx = Bz * Tz * Dz;
    f32_to_bf16_kernel<<<dim3((nx + 255) / 256), dim3(256), 0, stream>>>(x, xbf, nx);
    long nw0 = 4 * Hz * K0;
    pack_weights_kernel<<<dim3((nw0 + 255) / 256), dim3(256), 0, stream>>>(W0, W0p, (int)K0, nw0);
    long nw1 = 4 * Hz * K1;
    pack_weights_kernel<<<dim3((nw1 + 255) / 256), dim3(256), 0, stream>>>(W1, W1p, (int)K1, nw1);
    zero_bf16_kernel<<<dim3(8), dim3(256), 0, stream>>>(zb, 2048);
    long nc = Bz * Hz;
    zero_f32_kernel<<<dim3((nc + 255) / 256), dim3(256), 0, stream>>>(c0, nc);
    zero_f32_kernel<<<dim3((nc + 255) / 256), dim3(256), 0, stream>>>(c1, nc);
  }

  const dim3 grid(LSTM_HDIM / 16);   // 64 blocks
  const dim3 blk(256);               // 8 wave32 waves

  // ---- layer 0: h sequence goes straight into y0[B,T,H] (bf16) ----
  for (int t = 0; t < (int)Tz; ++t) {
    const bf16_t* hp = (t == 0) ? zb : (y0 + (long)(t - 1) * Hz);
    long shp = (t == 0) ? 0 : Tz * Hz;
    lstm_step_kernel<LSTM_DIN0><<<grid, blk, 0, stream>>>(
        xbf + (long)t * Dz, Tz * Dz,       // x_t rows, stride T*D
        hp, shp,                           // h_{t-1}
        W0p, b0, c0,
        y0 + (long)t * Hz, Tz * Hz,        // h_t -> y0[:, t, :]
        nullptr, 0);
  }

  // ---- layer 1: reads y0[:, t, :], ping-pongs bf16 h, writes fp32 out ----
  for (int t = 0; t < (int)Tz; ++t) {
    const bf16_t* hp = (t == 0) ? zb : (((t - 1) & 1) ? h1b : h1a);
    long shp = (t == 0) ? 0 : Hz;
    bf16_t* ho = (t & 1) ? h1b : h1a;
    lstm_step_kernel<LSTM_HDIM><<<grid, blk, 0, stream>>>(
        y0 + (long)t * Hz, Tz * Hz,        // layer-0 output at step t
        hp, shp,
        W1p, b1, c1,
        ho, Hz,                            // bf16 h for next step
        out + (long)t * Hz, Tz * Hz);      // fp32 h -> out[:, t, :]
  }
}
